// QuantCrossAttention_61761629716754
// MI455X (gfx1250) — 16-bit and 8-bit matrix operand order repaired and checked by reading, compiled, not run on hardware
//
#include <hip/hip_runtime.h>

typedef _Float16 h16;
typedef __attribute__((ext_vector_type(8)))  _Float16 v8h;
typedef __attribute__((ext_vector_type(16))) _Float16 v16h;
typedef __attribute__((ext_vector_type(8)))  float    v8f;
typedef __attribute__((ext_vector_type(8)))  int      v8i;

#define BB   2
#define NQ   4096
#define MK   1024
#define DD   512
#define DC   768
#define HH   8
#define DH   64
#define SCAL_SCALE 0.125f

static __device__ __forceinline__ v8f vz8f() {
  v8f z;
#pragma unroll
  for (int i = 0; i < 8; ++i) z[i] = 0.f;
  return z;
}
static __device__ __forceinline__ v8i vz8i() {
  v8i z;
#pragma unroll
  for (int i = 0; i < 8; ++i) z[i] = 0;
  return z;
}

__global__ void k_init(unsigned* scal) {
  if (threadIdx.x == 0) {
    scal[0] = 0u;
    scal[1] = 0u;
    scal[2] = 0u;
    scal[3] = 0x7F800000u;
  }
}

__global__ void k_cvt_f16(const float* __restrict__ src, h16* __restrict__ dst, int n) {
  int i = blockIdx.x * 256 + threadIdx.x;
  if (i < n) dst[i] = (h16)src[i];
}

__global__ void k_wt_f16(const float* __restrict__ W, h16* __restrict__ Wt, int K, int N) {
  int i = blockIdx.x * 256 + threadIdx.x;
  if (i < K * N) {
    int k = i / N, n = i - k * N;
    Wt[(size_t)n * K + k] = (h16)W[i];
  }
}

__global__ __launch_bounds__(256)
void k_gemm_f16(const h16* __restrict__ A, const h16* __restrict__ Bt,
                float* __restrict__ C, const float* __restrict__ bias,
                unsigned* __restrict__ maxslot, int M, int N, int K) {
  const int lane = threadIdx.x & 31;
  const int wave = threadIdx.x >> 5;
  const int half = lane >> 4;
  const int l16  = lane & 15;
  const int m0 = blockIdx.x * 128 + wave * 16;
  const int n0 = blockIdx.y * 64;

  v8f acc[4];
#pragma unroll
  for (int t = 0; t < 4; ++t) acc[t] = vz8f();

  const h16* arow = A + (size_t)(m0 + l16) * K;
  for (int kk = 0; kk < K; kk += 32) {
    v16h af;
    {
      v8h a0 = *(const v8h*)(arow + kk + half * 16);
      v8h a1 = *(const v8h*)(arow + kk + half * 16 + 8);
#pragma unroll
      for (int i = 0; i < 8; ++i) { af[i] = a0[i]; af[i + 8] = a1[i]; }
    }
#pragma unroll
    for (int t = 0; t < 4; ++t) {
      v16h bf = *(const v16h*)(Bt + (size_t)(n0 + t * 16 + l16) * K + kk + half * 16);
      acc[t] = __builtin_amdgcn_wmma_f32_16x16x32_f16(false, af, false, bf,
                                                      (short)0, acc[t], false, false);
    }
  }

  float mx = 0.f;
#pragma unroll
  for (int t = 0; t < 4; ++t) {
    int col = n0 + t * 16 + l16;
    float bv = bias ? bias[col] : 0.f;
#pragma unroll
    for (int r = 0; r < 8; ++r) {
      float v = acc[t][r] + bv;
      mx = fmaxf(mx, fabsf(v));
      C[(size_t)(m0 + r + half * 8) * N + col] = v;
    }
  }
  if (maxslot) {
#pragma unroll
    for (int s = 16; s >= 1; s >>= 1) mx = fmaxf(mx, __shfl_xor(mx, s, 32));
    if (lane == 0) atomicMax(maxslot, __float_as_uint(mx));
  }
}

__global__ void k_quant(const float* __restrict__ src, signed char* __restrict__ dst,
                        const unsigned* __restrict__ scal, int slot, int n) {
  int i = blockIdx.x * 256 + threadIdx.x;
  if (i >= n) return;
  float delta = __uint_as_float(scal[slot]) * (1.0f / 127.0f);
  float inv = delta > 0.f ? 1.0f / delta : 0.f;
  int q = __float2int_rn(src[i] * inv);
  q = q < -128 ? -128 : (q > 127 ? 127 : q);
  dst[i] = (signed char)q;
}

__global__ void k_quant_v_t(const float* __restrict__ v, signed char* __restrict__ v8t,
                            const unsigned* __restrict__ scal) {
  int i = blockIdx.x * 256 + threadIdx.x;
  if (i >= BB * DD * MK) return;
  int j   = i & (MK - 1);
  int col = (i >> 10) & (DD - 1);
  int b   = i >> 19;
  float delta = __uint_as_float(scal[2]) * (1.0f / 127.0f);
  float inv = delta > 0.f ? 1.0f / delta : 0.f;
  float x = v[((size_t)(b * MK + j)) * DD + col];
  int q = __float2int_rn(x * inv);
  q = q < -128 ? -128 : (q > 127 ? 127 : q);
  int hh = col >> 6, nn = col & 63;
  v8t[(((size_t)(b * HH + hh)) * DH + nn) * MK + j] = (signed char)q;
}

static __device__ __forceinline__ v8i load_a_iu8(const signed char* row, int half) {
  int koff = half * 16;
  uint2 c0 = *(const uint2*)(row + koff);
  uint2 c1 = *(const uint2*)(row + 8 + koff);
  uint2 c2 = *(const uint2*)(row + 32 + koff);
  uint2 c3 = *(const uint2*)(row + 40 + koff);
  v8i a;
  a[0] = (int)c0.x; a[1] = (int)c0.y; a[2] = (int)c1.x; a[3] = (int)c1.y;
  a[4] = (int)c2.x; a[5] = (int)c2.y; a[6] = (int)c3.x; a[7] = (int)c3.y;
  return a;
}
static __device__ __forceinline__ v8i load_b_iu8(const signed char* col, int half) {
  uint4 d0 = *(const uint4*)(col + half * 16);
  uint4 d1 = *(const uint4*)(col + 32 + half * 16);
  v8i b;
  b[0] = (int)d0.x; b[1] = (int)d0.y; b[2] = (int)d0.z; b[3] = (int)d0.w;
  b[4] = (int)d1.x; b[5] = (int)d1.y; b[6] = (int)d1.z; b[7] = (int)d1.w;
  return b;
}

__global__ __launch_bounds__(128)
void k_attn_stats(const signed char* __restrict__ q8, const signed char* __restrict__ k8,
                  float* __restrict__ mbuf, float* __restrict__ lbuf,
                  unsigned* __restrict__ scal) {
  const int lane = threadIdx.x & 31, wave = threadIdx.x >> 5;
  const int half = lane >> 4, l16 = lane & 15;
  const int bh = blockIdx.y, b = bh >> 3, hh = bh & 7;
  const int row0 = blockIdx.x * 64 + wave * 16;
  const float dq = __uint_as_float(scal[0]) * (1.0f / 127.0f);
  const float dk = __uint_as_float(scal[1]) * (1.0f / 127.0f);
  const float sscale = dq * dk * SCAL_SCALE;

  const signed char* qrow = q8 + ((size_t)(b * NQ + row0 + l16)) * DD + hh * DH;
  v8i af = load_a_iu8(qrow, half);

  float mrow[8], lrow[8];
#pragma unroll
  for (int r = 0; r < 8; ++r) { mrow[r] = -INFINITY; lrow[r] = 0.f; }

  for (int j0 = 0; j0 < MK; j0 += 16) {
    const signed char* krow = k8 + ((size_t)(b * MK + j0 + l16)) * DD + hh * DH;
    v8i bf = load_b_iu8(krow, half);
    v8i sim = __builtin_amdgcn_wmma_i32_16x16x64_iu8(true, af, true, bf, vz8i(), false, false);
#pragma unroll
    for (int r = 0; r < 8; ++r) {
      float s = (float)sim[r] * sscale;
      float tm = s;
#pragma unroll
      for (int sft = 8; sft >= 1; sft >>= 1) tm = fmaxf(tm, __shfl_xor(tm, sft, 32));
      float mnew = fmaxf(mrow[r], tm);
      float e = __expf(s - mnew);
#pragma unroll
      for (int sft = 8; sft >= 1; sft >>= 1) e += __shfl_xor(e, sft, 32);
      lrow[r] = lrow[r] * __expf(mrow[r] - mnew) + e;
      mrow[r] = mnew;
    }
  }
  if (l16 == 0) {
#pragma unroll
    for (int r = 0; r < 8; ++r) {
      int row = row0 + r + half * 8;
      mbuf[(size_t)bh * NQ + row] = mrow[r];
      lbuf[(size_t)bh * NQ + row] = lrow[r];
      atomicMin(scal + 3, __float_as_uint(lrow[r]));
    }
  }
}

__global__ __launch_bounds__(128)
void k_attn_out(const signed char* __restrict__ q8, const signed char* __restrict__ k8,
                const signed char* __restrict__ v8t,
                const float* __restrict__ mbuf, const float* __restrict__ lbuf,
                const unsigned* __restrict__ scal, h16* __restrict__ o16) {
  __shared__ __align__(16) unsigned char a8s[4][16][64];

  const int lane = threadIdx.x & 31, wave = threadIdx.x >> 5;
  const int half = lane >> 4, l16 = lane & 15;
  const int bh = blockIdx.y, b = bh >> 3, hh = bh & 7;
  const int row0 = blockIdx.x * 64 + wave * 16;

  const float dq = __uint_as_float(scal[0]) * (1.0f / 127.0f);
  const float dk = __uint_as_float(scal[1]) * (1.0f / 127.0f);
  const float dv = __uint_as_float(scal[2]) * (1.0f / 127.0f);
  const float sscale = dq * dk * SCAL_SCALE;
  const float lmin = __uint_as_float(scal[3]);
  const float da = (1.0f / lmin) * (1.0f / 127.0f);

  const signed char* qrow = q8 + ((size_t)(b * NQ + row0 + l16)) * DD + hh * DH;
  v8i af = load_a_iu8(qrow, half);

  float mrw[8], invlda[8];
#pragma unroll
  for (int r = 0; r < 8; ++r) {
    int row = row0 + r + half * 8;
    mrw[r] = mbuf[(size_t)bh * NQ + row];
    float l = lbuf[(size_t)bh * NQ + row];
    invlda[r] = 1.0f / (l * da);
  }

  v8i oacc[4];
#pragma unroll
  for (int t = 0; t < 4; ++t) oacc[t] = vz8i();

  for (int j0 = 0; j0 < MK; j0 += 64) {
#pragma unroll
    for (int ss = 0; ss < 4; ++ss) {
      int jj = j0 + ss * 16;
      const signed char* krow = k8 + ((size_t)(b * MK + jj + l16)) * DD + hh * DH;
      v8i bf = load_b_iu8(krow, half);
      v8i sim = __builtin_amdgcn_wmma_i32_16x16x64_iu8(true, af, true, bf, vz8i(), false, false);
#pragma unroll
      for (int r = 0; r < 8; ++r) {
        float s = (float)sim[r] * sscale;
        float p = __expf(s - mrw[r]);
        int q = __float2int_rn(p * invlda[r]);
        q = q < 0 ? 0 : (q > 127 ? 127 : q);
        a8s[wave][r + half * 8][ss * 16 + l16] = (unsigned char)q;
      }
    }
    __syncthreads();
    v8i aq = load_a_iu8((const signed char*)&a8s[wave][l16][0], half);
#pragma unroll
    for (int t = 0; t < 4; ++t) {
      const signed char* vcol = v8t + (((size_t)bh) * DH + t * 16 + l16) * MK + j0;
      v8i bf = load_b_iu8(vcol, half);
      oacc[t] = __builtin_amdgcn_wmma_i32_16x16x64_iu8(false, aq, true, bf, oacc[t], false, false);
    }
    __syncthreads();
  }

  const float osc = da * dv;
#pragma unroll
  for (int t = 0; t < 4; ++t) {
#pragma unroll
    for (int r = 0; r < 8; ++r) {
      int row = row0 + r + half * 8;
      int col = hh * DH + t * 16 + l16;
      o16[((size_t)(b * NQ + row)) * DD + col] = (h16)((float)oacc[t][r] * osc);
    }
  }
}

extern "C" void kernel_launch(void* const* d_in, const int* in_sizes, int n_in,
                              void* d_out, int out_size, void* d_ws, size_t ws_size,
                              hipStream_t stream) {
  const float* x   = (const float*)d_in[0];
  const float* ctx = (const float*)d_in[1];
  const float* Wq  = (const float*)d_in[2];
  const float* Wk  = (const float*)d_in[3];
  const float* Wv  = (const float*)d_in[4];
  const float* Wo  = (const float*)d_in[5];
  const float* bo  = (const float*)d_in[6];

  char* base = (char*)d_ws;
  size_t off = 0;
  auto alloc = [&](size_t bytes) { size_t o = off; off = (off + bytes + 255) & ~(size_t)255; return o; };

  unsigned* scal = (unsigned*)(base + alloc(16));
  float* qf  = (float*)(base + alloc((size_t)BB * NQ * DD * 4));
  float* kf  = (float*)(base + alloc((size_t)BB * MK * DD * 4));
  float* vf  = (float*)(base + alloc((size_t)BB * MK * DD * 4));
  h16*  x16  = (h16*) (base + alloc((size_t)BB * NQ * DD * 2));
  h16*  c16  = (h16*) (base + alloc((size_t)BB * MK * DC * 2));
  h16*  wq16 = (h16*) (base + alloc((size_t)DD * DD * 2));
  h16*  wk16 = (h16*) (base + alloc((size_t)DD * DC * 2));
  h16*  wv16 = (h16*) (base + alloc((size_t)DD * DC * 2));
  h16*  wo16 = (h16*) (base + alloc((size_t)DD * DD * 2));
  signed char* q8  = (signed char*)(base + alloc((size_t)BB * NQ * DD));
  signed char* k8  = (signed char*)(base + alloc((size_t)BB * MK * DD));
  signed char* v8t = (signed char*)(base + alloc((size_t)BB * HH * DH * MK));
  h16*  o16  = (h16*) (base + alloc((size_t)BB * NQ * DD * 2));
  float* mbuf = (float*)(base + alloc((size_t)BB * HH * NQ * 4));
  float* lbuf = (float*)(base + alloc((size_t)BB * HH * NQ * 4));

  auto cdiv = [](int a, int b) { return (a + b - 1) / b; };

  k_init<<<1, 32, 0, stream>>>(scal);

  k_cvt_f16<<<cdiv(BB * NQ * DD, 256), 256, 0, stream>>>(x,   x16, BB * NQ * DD);
  k_cvt_f16<<<cdiv(BB * MK * DC, 256), 256, 0, stream>>>(ctx, c16, BB * MK * DC);
  k_wt_f16<<<cdiv(DD * DD, 256), 256, 0, stream>>>(Wq, wq16, DD, DD);
  k_wt_f16<<<cdiv(DC * DD, 256), 256, 0, stream>>>(Wk, wk16, DC, DD);
  k_wt_f16<<<cdiv(DC * DD, 256), 256, 0, stream>>>(Wv, wv16, DC, DD);
  k_wt_f16<<<cdiv(DD * DD, 256), 256, 0, stream>>>(Wo, wo16, DD, DD);

  dim3 gq(BB * NQ / 128, DD / 64);
  dim3 gkv(BB * MK / 128, DD / 64);
  k_gemm_f16<<<gq,  256, 0, stream>>>(x16, wq16, qf, nullptr, scal + 0, BB * NQ, DD, DD);
  k_gemm_f16<<<gkv, 256, 0, stream>>>(c16, wk16, kf, nullptr, scal + 1, BB * MK, DD, DC);
  k_gemm_f16<<<gkv, 256, 0, stream>>>(c16, wv16, vf, nullptr, scal + 2, BB * MK, DD, DC);

  k_quant<<<cdiv(BB * NQ * DD, 256), 256, 0, stream>>>(qf, q8, scal, 0, BB * NQ * DD);
  k_quant<<<cdiv(BB * MK * DD, 256), 256, 0, stream>>>(kf, k8, scal, 1, BB * MK * DD);
  k_quant_v_t<<<cdiv(BB * DD * MK, 256), 256, 0, stream>>>(vf, v8t, scal);

  dim3 ga(NQ / 64, BB * HH);
  k_attn_stats<<<ga, 128, 0, stream>>>(q8, k8, mbuf, lbuf, scal);
  k_attn_out<<<ga, 128, 0, stream>>>(q8, k8, v8t, mbuf, lbuf, scal, o16);

  k_gemm_f16<<<gq, 256, 0, stream>>>(o16, wo16, (float*)d_out, bo, nullptr, BB * NQ, DD, DD);
}
